// MultiHeadLinearAttention_71193377899269
// MI455X (gfx1250) — matrix operand order repaired and checked by reading, not compiled, not run on hardware
//
#include <hip/hip_runtime.h>
#include <hip/hip_bf16.h>
#include <stdint.h>

// Problem constants (match reference)
#define D_MODEL 512
#define S_LEN   8192
#define BATCH   4
#define HEADS   8
#define DK      64
#define ROWS    (BATCH * S_LEN)   // 32768
#define NCHUNK  8                 // S chunks for kv partial reduction
#define CHUNK   (S_LEN / NCHUNK)  // 1024

typedef __attribute__((ext_vector_type(16))) __bf16 v16bf;
typedef __attribute__((ext_vector_type(8)))  float  v8f;

// ---- CDNA5 async global->LDS copy (GLOBAL_LOAD_ASYNC_TO_LDS_B128, ASYNCcnt) ----
#if defined(__gfx1250__) && __has_builtin(__builtin_amdgcn_global_load_async_to_lds_b128)
#define HAVE_ASYNC_LDS 1
#else
#define HAVE_ASYNC_LDS 0
#endif

#if HAVE_ASYNC_LDS
typedef int i32x4_ __attribute__((vector_size(16)));
#define CP16B(src, dst)                                                     \
  __builtin_amdgcn_global_load_async_to_lds_b128(                           \
      (__attribute__((address_space(1))) i32x4_*)                           \
          (__attribute__((address_space(1))) void*)(src),                   \
      (__attribute__((address_space(3))) i32x4_*)                           \
          (__attribute__((address_space(3))) void*)(dst),                   \
      0, 0)
#if __has_builtin(__builtin_amdgcn_s_wait_asynccnt)
#define WAIT_ASYNC(n) __builtin_amdgcn_s_wait_asynccnt(n)
#else
#define WAIT_ASYNC(n) asm volatile("s_wait_asynccnt %0" ::"n"(n))
#endif
#else
#define CP16B(src, dst) (*(uint4*)(dst) = *(const uint4*)(src))
#define WAIT_ASYNC(n)
#endif

__device__ __forceinline__ unsigned short f2bf(float f) {
  union { float f; unsigned u; } a; a.f = f;
  unsigned u = a.u;
  u += 0x7FFFu + ((u >> 16) & 1u);   // round-to-nearest-even
  return (unsigned short)(u >> 16);
}
__device__ __forceinline__ float bf2f(unsigned short h) {
  union { unsigned u; float f; } a; a.u = ((unsigned)h) << 16;
  return a.f;
}

union Frag {
  v16bf v;
  unsigned short h[16];
  uint4 q[2];
};
union U4 { uint4 q; unsigned short h[8]; };

// ---------------------------------------------------------------------------
// Kernel 0: convert one [512 x 512] f32 weight (row = out e, col = in d) into
// bf16 fragments pre-swizzled for WMMA B operand of Y = X * W^T.
// Fragment f = nt*16 + kt covers K rows [32kt,32kt+32), N cols [16nt,16nt+16).
// Lane l stores halves for n = l&15, k = 32kt+(l>>4)*8+{0..7} then +16+{0..7}.
// ---------------------------------------------------------------------------
__global__ __launch_bounds__(256) void prep_weights(
    const float* __restrict__ W, unsigned short* __restrict__ swz)
{
  int gid  = blockIdx.x * 256 + threadIdx.x;  // 0..16383
  int lane = gid & 31;
  int f    = gid >> 5;                        // 0..511
  int nt   = f >> 4;
  int kt   = f & 15;
  int e     = nt * 16 + (lane & 15);
  int dbase = kt * 32 + (lane >> 4) * 8;
  unsigned short* o = swz + ((size_t)f * 32 + lane) * 16;
  const float* wrow = W + (size_t)e * D_MODEL + dbase;
#pragma unroll
  for (int j = 0; j < 16; ++j) o[j] = f2bf(wrow[j < 8 ? j : j + 8]);
}

// ---------------------------------------------------------------------------
// Kernel 1: fused projection GEMM (triple-buffered async LDS staging + WMMA).
//   G1 = X @ W1^T + b1 ; G2 = X @ W2^T + b2 ; Y = silu(G1) * G2
//   PHI=1 additionally applies phi(y) = elu(y)+1 (q/k paths).
// Block = 128 threads (4 waves). Block tile M=64, N=64. Wave tile 16x64.
// grid = (ROWS/64, 512/64)
// ---------------------------------------------------------------------------
template <int PHI>
__global__ __launch_bounds__(128) void proj_gemm(
    const float* __restrict__ X,
    const unsigned short* __restrict__ W1s,
    const unsigned short* __restrict__ W2s,
    const float* __restrict__ b1,
    const float* __restrict__ b2,
    unsigned short* __restrict__ Y)
{
  __shared__ __align__(16) unsigned short ldsB[3][8 * 512];  // 24KB

  const int t      = threadIdx.x;
  const int lane   = t & 31;
  const int wid    = t >> 5;
  const int laneN  = lane & 15;
  const int laneHi = lane >> 4;
  const int m0     = blockIdx.x * 64 + wid * 16;
  const int nblk   = blockIdx.y;              // N cols [nblk*64, +64)

  v8f acc1[4] = {}, acc2[4] = {};
  const float* xrow = X + (size_t)(m0 + laneN) * D_MODEL + laneHi * 8;

  // stage the 8 B fragments (4 n-tiles x {W1,W2}) of K-step kt into ldsB[buf]
  auto stageB = [&](int buf, int kt) {
#pragma unroll
    for (int c = 0; c < 4; ++c) {
      int id  = c * 128 + t;     // 0..511 : 512 x 16B segments = 8KB
      int j   = id >> 6;         // fragment 0..7 (0-3 -> W1, 4-7 -> W2)
      int seg = id & 63;
      const unsigned short* src =
          (j < 4 ? W1s : W2s) +
          ((size_t)((nblk * 4 + (j & 3)) * 16 + kt) * 512 + seg * 8);
      unsigned short* dst = &ldsB[buf][j * 512 + seg * 8];
      CP16B(src, dst);
    }
  };

  // prologue: stage kt=0 and kt=1, load raw A(0)
  stageB(0, 0);
  stageB(1, 1);
  float4 ar[4];
  {
    const float* xp = xrow;
    ar[0] = *(const float4*)(xp + 0);
    ar[1] = *(const float4*)(xp + 4);
    ar[2] = *(const float4*)(xp + 16);
    ar[3] = *(const float4*)(xp + 20);
  }

  for (int kt = 0; kt < 16; ++kt) {
    // ensure ldsB[kt%3] (staged 2 iterations ago) is complete, then barrier
    if (kt < 15) { WAIT_ASYNC(4); } else { WAIT_ASYNC(0); }
    __syncthreads();

    const int buf = kt % 3;

    // kick off stage for kt+2 (overlaps with this step's DS reads + WMMA)
    if (kt + 2 < 16) stageB((kt + 2) % 3, kt + 2);

    // convert current A fragment
    float fl[16] = {ar[0].x, ar[0].y, ar[0].z, ar[0].w,
                    ar[1].x, ar[1].y, ar[1].z, ar[1].w,
                    ar[2].x, ar[2].y, ar[2].z, ar[2].w,
                    ar[3].x, ar[3].y, ar[3].z, ar[3].w};
    Frag a;
#pragma unroll
    for (int i = 0; i < 16; ++i) a.h[i] = f2bf(fl[i]);

    // preload raw A for next K-step (waited only at next iteration's barrier)
    if (kt + 1 < 16) {
      const float* xp = xrow + (kt + 1) * 32;
      ar[0] = *(const float4*)(xp + 0);
      ar[1] = *(const float4*)(xp + 4);
      ar[2] = *(const float4*)(xp + 16);
      ar[3] = *(const float4*)(xp + 20);
    }

#pragma unroll
    for (int j = 0; j < 4; ++j) {
      const unsigned short* lb1 = &ldsB[buf][j * 512 + lane * 16];
      const unsigned short* lb2 = &ldsB[buf][(j + 4) * 512 + lane * 16];
      Frag bw1, bw2;
      bw1.q[0] = *(const uint4*)(lb1);
      bw1.q[1] = *(const uint4*)(lb1 + 8);
      bw2.q[0] = *(const uint4*)(lb2);
      bw2.q[1] = *(const uint4*)(lb2 + 8);
      acc1[j] = __builtin_amdgcn_wmma_f32_16x16x32_bf16(
          false, a.v, false, bw1.v, (short)0, acc1[j], false, false);
      acc2[j] = __builtin_amdgcn_wmma_f32_16x16x32_bf16(
          false, a.v, false, bw2.v, (short)0, acc2[j], false, false);
    }
  }

  // ---- epilogue: bias, silu*gate, optional phi, store bf16 ----
#pragma unroll
  for (int j = 0; j < 4; ++j) {
    int e = nblk * 64 + j * 16 + laneN;
    float bb1 = b1[e];
    float bb2 = b2[e];
#pragma unroll
    for (int r = 0; r < 8; ++r) {
      int m = m0 + r + laneHi * 8;
      float g1 = acc1[j][r] + bb1;
      float g2 = acc2[j][r] + bb2;
      float y = (g1 / (1.0f + __expf(-g1))) * g2;        // silu(g1)*g2
      if (PHI) y = (y > 0.0f) ? (y + 1.0f) : __expf(y);  // elu(y)+1
      Y[(size_t)m * D_MODEL + e] = f2bf(y);
    }
  }
}

// ---------------------------------------------------------------------------
// Kernel 2: per-(b,h,chunk) partial reduction
//   kv[d][e] += phi_k[s][d] * v[s][e] ; ks[d] += phi_k[s][d]
// grid = (B*H, NCHUNK), block = 256. Thread owns d = t>>2, e in [(t&3)*16,+16).
// ---------------------------------------------------------------------------
__global__ __launch_bounds__(256) void kv_partial(
    const unsigned short* __restrict__ PhiK,
    const unsigned short* __restrict__ Vp,
    float* __restrict__ kvp, float* __restrict__ ksp)
{
  int bh = blockIdx.x;
  int b = bh >> 3, h = bh & 7;
  int chunk = blockIdx.y;
  int t = threadIdx.x;
  int d  = t >> 2;
  int eg = (t & 3) * 16;

  __shared__ __align__(16) unsigned short lk[32 * 64];
  __shared__ __align__(16) unsigned short lv[32 * 64];

  float acc[16] = {};
  float ksacc = 0.f;
  size_t rowbase = ((size_t)b * S_LEN + (size_t)chunk * CHUNK) * D_MODEL + h * DK;

  for (int s0 = 0; s0 < CHUNK; s0 += 32) {
    int li = t * 8;               // 2048 halves / 256 threads
    int sr = li >> 6;
    int sc = li & 63;
    size_t g = rowbase + (size_t)(s0 + sr) * D_MODEL + sc;
    *(uint4*)(lk + li) = *(const uint4*)(PhiK + g);
    *(uint4*)(lv + li) = *(const uint4*)(Vp + g);
    __syncthreads();
#pragma unroll 4
    for (int ss = 0; ss < 32; ++ss) {
      float pk = bf2f(lk[ss * 64 + d]);
      ksacc += pk;
#pragma unroll
      for (int i = 0; i < 16; ++i)
        acc[i] += pk * bf2f(lv[ss * 64 + eg + i]);
    }
    __syncthreads();
  }

  float* kvo = kvp + ((size_t)chunk * (BATCH * HEADS) + bh) * (DK * DK);
#pragma unroll
  for (int i = 0; i < 16; ++i) kvo[d * DK + eg + i] = acc[i];
  if ((t & 3) == 0)
    ksp[((size_t)chunk * (BATCH * HEADS) + bh) * DK + d] = ksacc;
}

// Kernel 2b: sum partial chunks. grid = B*H, block = 256.
__global__ __launch_bounds__(256) void kv_reduce(
    const float* __restrict__ kvp, const float* __restrict__ ksp,
    float* __restrict__ kv, float* __restrict__ ks)
{
  int bh = blockIdx.x;
  int t = threadIdx.x;
#pragma unroll
  for (int i = 0; i < 16; ++i) {
    int idx = t * 16 + i;
    float s = 0.f;
    for (int c = 0; c < NCHUNK; ++c)
      s += kvp[((size_t)c * (BATCH * HEADS) + bh) * (DK * DK) + idx];
    kv[(size_t)bh * (DK * DK) + idx] = s;
  }
  if (t < DK) {
    float s = 0.f;
    for (int c = 0; c < NCHUNK; ++c)
      s += ksp[((size_t)c * (BATCH * HEADS) + bh) * DK + t];
    ks[(size_t)bh * DK + t] = s;
  }
}

// ---------------------------------------------------------------------------
// Kernel 3: attn[s][e] = (phi_q[s,:] @ kv[:,e]) / (phi_q[s,:].ks + eps)
// grid = (B*H, S/64), block = 256; 4 threads per s-row, 16 e each.
// ---------------------------------------------------------------------------
__global__ __launch_bounds__(256) void attn_kernel(
    const unsigned short* __restrict__ PhiQ,
    const float* __restrict__ kv, const float* __restrict__ ks,
    unsigned short* __restrict__ Attn)
{
  int bh = blockIdx.x;
  int b = bh >> 3, h = bh & 7;
  int t = threadIdx.x;

  __shared__ __align__(16) float lkv[DK * DK];
  __shared__ float lks[DK];
  for (int i = t; i < DK * DK; i += 256) lkv[i] = kv[(size_t)bh * (DK * DK) + i];
  if (t < DK) lks[t] = ks[(size_t)bh * DK + t];
  __syncthreads();

  int srow = blockIdx.y * 64 + (t >> 2);
  int eg   = (t & 3) * 16;
  size_t base = ((size_t)b * S_LEN + srow) * D_MODEL + h * DK;

  const float4* lkv4 = (const float4*)lkv;
  float qks = 0.f;
  float num[16] = {};
  for (int d0 = 0; d0 < DK; d0 += 8) {
    U4 u; u.q = *(const uint4*)(PhiQ + base + d0);
#pragma unroll
    for (int j = 0; j < 8; ++j) {
      float pq = bf2f(u.h[j]);
      int d = d0 + j;
      qks += pq * lks[d];
#pragma unroll
      for (int i = 0; i < 4; ++i) {
        float4 kvv = lkv4[(d * DK + eg) / 4 + i];
        num[4 * i + 0] += pq * kvv.x;
        num[4 * i + 1] += pq * kvv.y;
        num[4 * i + 2] += pq * kvv.z;
        num[4 * i + 3] += pq * kvv.w;
      }
    }
  }
  float inv = 1.0f / (qks + 1e-6f);
  U4 o0, o1;
#pragma unroll
  for (int i = 0; i < 8; ++i) {
    o0.h[i] = f2bf(num[i] * inv);
    o1.h[i] = f2bf(num[8 + i] * inv);
  }
  *(uint4*)(Attn + base + eg)     = o0.q;
  *(uint4*)(Attn + base + eg + 8) = o1.q;
}

// ---------------------------------------------------------------------------
// Kernel 4: final projection  out = Attn @ out_w^T + out_b  (f32 output)
// Same triple-buffered staging; A already bf16.
// ---------------------------------------------------------------------------
__global__ __launch_bounds__(128) void out_gemm(
    const unsigned short* __restrict__ Xbf,
    const unsigned short* __restrict__ Ws,
    const float* __restrict__ bias,
    float* __restrict__ Y)
{
  __shared__ __align__(16) unsigned short ldsB[3][4 * 512];  // 12KB

  const int t      = threadIdx.x;
  const int lane   = t & 31;
  const int wid    = t >> 5;
  const int laneN  = lane & 15;
  const int laneHi = lane >> 4;
  const int m0     = blockIdx.x * 64 + wid * 16;
  const int nblk   = blockIdx.y;

  v8f acc[4] = {};
  const unsigned short* xrow = Xbf + (size_t)(m0 + laneN) * D_MODEL + laneHi * 8;

  auto stageB = [&](int buf, int kt) {
#pragma unroll
    for (int c = 0; c < 2; ++c) {
      int id  = c * 128 + t;     // 0..255 : 256 x 16B = 4KB
      int j   = id >> 6;         // fragment 0..3
      int seg = id & 63;
      const unsigned short* src =
          Ws + ((size_t)((nblk * 4 + j) * 16 + kt) * 512 + seg * 8);
      unsigned short* dst = &ldsB[buf][j * 512 + seg * 8];
      CP16B(src, dst);
    }
  };

  stageB(0, 0);
  stageB(1, 1);
  U4 aq0, aq1;
  aq0.q = *(const uint4*)(xrow);
  aq1.q = *(const uint4*)(xrow + 16);

  for (int kt = 0; kt < 16; ++kt) {
    if (kt < 15) { WAIT_ASYNC(2); } else { WAIT_ASYNC(0); }
    __syncthreads();

    const int buf = kt % 3;
    if (kt + 2 < 16) stageB((kt + 2) % 3, kt + 2);

    Frag a;
    a.q[0] = aq0.q;
    a.q[1] = aq1.q;
    if (kt + 1 < 16) {
      aq0.q = *(const uint4*)(xrow + (kt + 1) * 32);
      aq1.q = *(const uint4*)(xrow + (kt + 1) * 32 + 16);
    }

#pragma unroll
    for (int j = 0; j < 4; ++j) {
      const unsigned short* lb = &ldsB[buf][j * 512 + lane * 16];
      Frag bw;
      bw.q[0] = *(const uint4*)(lb);
      bw.q[1] = *(const uint4*)(lb + 8);
      acc[j] = __builtin_amdgcn_wmma_f32_16x16x32_bf16(
          false, a.v, false, bw.v, (short)0, acc[j], false, false);
    }
  }

#pragma unroll
  for (int j = 0; j < 4; ++j) {
    int e = nblk * 64 + j * 16 + laneN;
    float bb = bias[e];
#pragma unroll
    for (int r = 0; r < 8; ++r) {
      int m = m0 + r + laneHi * 8;
      Y[(size_t)m * D_MODEL + e] = acc[j][r] + bb;
    }
  }
}

// ---------------------------------------------------------------------------
extern "C" void kernel_launch(void* const* d_in, const int* in_sizes, int n_in,
                              void* d_out, int out_size, void* d_ws, size_t ws_size,
                              hipStream_t stream)
{
  const float* query = (const float*)d_in[0];
  const float* keyt  = (const float*)d_in[1];
  const float* value = (const float*)d_in[2];
  const float* Wm[7];
  for (int i = 0; i < 7; ++i) Wm[i] = (const float*)d_in[3 + i];
  const float* q_b1 = (const float*)d_in[10];
  const float* q_b2 = (const float*)d_in[11];
  const float* k_b1 = (const float*)d_in[12];
  const float* k_b2 = (const float*)d_in[13];
  const float* v_b1 = (const float*)d_in[14];
  const float* v_b2 = (const float*)d_in[15];
  const float* out_b = (const float*)d_in[16];

  // workspace carve-up (all offsets 256B aligned)
  char* ws = (char*)d_ws;
  size_t off = 0;
  auto carve = [&](size_t bytes) {
    char* p = ws + off;
    off += (bytes + 255) & ~(size_t)255;
    return p;
  };
  const size_t WHALF = (size_t)D_MODEL * D_MODEL;           // 262144 halves
  unsigned short* swzW = (unsigned short*)carve(7 * WHALF * 2);
  unsigned short* phiQ = (unsigned short*)carve((size_t)ROWS * D_MODEL * 2);
  unsigned short* phiK = (unsigned short*)carve((size_t)ROWS * D_MODEL * 2);
  unsigned short* Vp   = (unsigned short*)carve((size_t)ROWS * D_MODEL * 2);
  unsigned short* Attn = (unsigned short*)carve((size_t)ROWS * D_MODEL * 2);
  float* kvp = (float*)carve((size_t)NCHUNK * BATCH * HEADS * DK * DK * 4);
  float* ksp = (float*)carve((size_t)NCHUNK * BATCH * HEADS * DK * 4);
  float* kv  = (float*)carve((size_t)BATCH * HEADS * DK * DK * 4);
  float* ks  = (float*)carve((size_t)BATCH * HEADS * DK * 4);

  // 0) weights -> swizzled bf16 fragments
  for (int i = 0; i < 7; ++i)
    prep_weights<<<64, 256, 0, stream>>>(Wm[i], swzW + (size_t)i * WHALF);

  // 1) fused projections (phi for q/k, plain for v)
  dim3 ggrid(ROWS / 64, D_MODEL / 64);
  proj_gemm<1><<<ggrid, 128, 0, stream>>>(query, swzW + 0 * WHALF, swzW + 1 * WHALF,
                                          q_b1, q_b2, phiQ);
  proj_gemm<1><<<ggrid, 128, 0, stream>>>(keyt,  swzW + 2 * WHALF, swzW + 3 * WHALF,
                                          k_b1, k_b2, phiK);
  proj_gemm<0><<<ggrid, 128, 0, stream>>>(value, swzW + 4 * WHALF, swzW + 5 * WHALF,
                                          v_b1, v_b2, Vp);

  // 2) key_value + key_sum reductions
  kv_partial<<<dim3(BATCH * HEADS, NCHUNK), 256, 0, stream>>>(phiK, Vp, kvp, ksp);
  kv_reduce<<<BATCH * HEADS, 256, 0, stream>>>(kvp, ksp, kv, ks);

  // 3) numerator / denominator -> attn (bf16, head-merged)
  attn_kernel<<<dim3(BATCH * HEADS, S_LEN / 64), 256, 0, stream>>>(phiQ, kv, ks, Attn);

  // 4) output projection -> f32 d_out
  out_gemm<<<ggrid, 128, 0, stream>>>(Attn, swzW + 6 * WHALF, out_b, (float*)d_out);
}
